// AttentionBlock_17497696764175
// MI455X (gfx1250) — matrix operand order repaired and checked by reading, not compiled, not run on hardware
//
#include <hip/hip_runtime.h>

typedef __attribute__((ext_vector_type(16))) _Float16 v16h;
typedef __attribute__((ext_vector_type(8)))  _Float16 v8h;
typedef __attribute__((ext_vector_type(8)))  float    v8f;

#define BB   4
#define CC   512
#define HWN  4096
#define NG   32
#define CPG  16
#define SCALE 0.04419417382415922f   // 1/sqrt(512)

// ---- WMMA helpers (CDNA5 wave32 layouts) ------------------------------------
// A fragment (16x32 f16, M x K): lane m=lane&15 holds row m; half h maps to
// K = 16g + h, g = lane>>4.
__device__ __forceinline__ v16h load_A_frag(const _Float16* base, int ld, int m, int g) {
    const _Float16* p = base + (size_t)m * ld + 16 * g;
    v8h lo = *(const v8h*)p;
    v8h hi = *(const v8h*)(p + 8);
    return __builtin_shufflevector(lo, hi, 0,1,2,3,4,5,6,7,8,9,10,11,12,13,14,15);
}

__device__ __forceinline__ v8f wmma_f16(v16h a, v16h b, v8f c) {
    return __builtin_amdgcn_wmma_f32_16x16x32_f16(false, a, false, b, (short)0, c,
                                                  false, false);
}

// online softmax row-stat update; D layout: M = r + 8g, N = lane&15 (16-lane groups)
__device__ __forceinline__ void online_update(v8f s, float* mrow, float* lrow) {
#pragma unroll
    for (int r = 0; r < 8; r++) {
        float v = s[r] * SCALE;
        float tmax = v;
        tmax = fmaxf(tmax, __shfl_xor(tmax, 1, 32));
        tmax = fmaxf(tmax, __shfl_xor(tmax, 2, 32));
        tmax = fmaxf(tmax, __shfl_xor(tmax, 4, 32));
        tmax = fmaxf(tmax, __shfl_xor(tmax, 8, 32));
        float mnew = fmaxf(mrow[r], tmax);
        float e = __expf(v - mnew);
        e += __shfl_xor(e, 1, 32);
        e += __shfl_xor(e, 2, 32);
        e += __shfl_xor(e, 4, 32);
        e += __shfl_xor(e, 8, 32);
        lrow[r] = lrow[r] * __expf(mrow[r] - mnew) + e;
        mrow[r] = mnew;
    }
}

// ---- 1. GroupNorm statistics ------------------------------------------------
__global__ void gn_stats(const float* __restrict__ x, float* __restrict__ meanr) {
    int bg = blockIdx.x;            // b*32 + g, 128 total
    int b = bg >> 5, g = bg & 31;
    const float* xb = x + (size_t)b * HWN * CC + g * CPG;
    float s = 0.f, sq = 0.f;
    for (int e = threadIdx.x; e < HWN * CPG; e += 256) {
        int pix = e >> 4, ch = e & 15;
        float v = xb[(size_t)pix * CC + ch];
        s += v; sq += v * v;
    }
    __shared__ float rs[256], rq[256];
    rs[threadIdx.x] = s; rq[threadIdx.x] = sq;
    __syncthreads();
    for (int st = 128; st > 0; st >>= 1) {
        if ((int)threadIdx.x < st) {
            rs[threadIdx.x] += rs[threadIdx.x + st];
            rq[threadIdx.x] += rq[threadIdx.x + st];
        }
        __syncthreads();
    }
    if (threadIdx.x == 0) {
        float mean = rs[0] * (1.f / 65536.f);
        float var  = rq[0] * (1.f / 65536.f) - mean * mean;
        meanr[2 * bg]     = mean;
        meanr[2 * bg + 1] = rsqrtf(var + 1e-6f);
    }
}

// ---- 2. Normalize + convert to f16 -----------------------------------------
__global__ void gn_norm(const float* __restrict__ x, const float* __restrict__ gamma,
                        const float* __restrict__ beta, const float* __restrict__ meanr,
                        _Float16* __restrict__ xn) {
    size_t i = (size_t)blockIdx.x * 256 + threadIdx.x;     // < B*HW*C
    int c  = (int)(i & (CC - 1));
    int bp = (int)(i >> 9);                                // b*HW + pix
    int b  = bp >> 12;
    int g  = c >> 4;
    float mean = meanr[2 * (b * NG + g)];
    float rstd = meanr[2 * (b * NG + g) + 1];
    float v = (x[i] - mean) * rstd * gamma[c] + beta[c];
    xn[i] = (_Float16)v;
}

// ---- 3. Weights: transpose + convert (wT[d][c] = w[c][d]) -------------------
__global__ void cvt_w(const float* __restrict__ wq, const float* __restrict__ wk,
                      const float* __restrict__ wv, const float* __restrict__ wo,
                      _Float16* __restrict__ wT) {
    int i = blockIdx.x * 256 + threadIdx.x;                // < 4*512*512
    int sel = i >> 18;
    int rem = i & (262144 - 1);
    int c = rem >> 9, d = rem & 511;
    const float* w = sel == 0 ? wq : sel == 1 ? wk : sel == 2 ? wv : wo;
    wT[(size_t)sel * 262144 + (size_t)d * 512 + c] = (_Float16)w[(size_t)c * 512 + d];
}

// ---- 4. Fused QKV GEMM: one wave = 32x16 tile (2 token tiles), 3 outputs ----
__global__ void qkv_gemm(const _Float16* __restrict__ xn, const _Float16* __restrict__ wT,
                         const float* __restrict__ bq, const float* __restrict__ bk,
                         const float* __restrict__ bv,
                         _Float16* __restrict__ qh, _Float16* __restrict__ kh,
                         _Float16* __restrict__ vT) {
    int wid  = blockIdx.x * 8 + (threadIdx.x >> 5);    // 16384 waves
    int lane = threadIdx.x & 31;
    int m = lane & 15, g = lane >> 4;
    int tn = wid & 31;               // out-channel tile
    int tp = (wid >> 5) & 127;       // token-pair tile (32 rows)
    int b  = wid >> 12;
    const _Float16* A = xn + ((size_t)b * HWN + tp * 32) * CC;
    v8f aq0 = {}, ak0 = {}, av0 = {}, aq1 = {}, ak1 = {}, av1 = {};
    for (int kk = 0; kk < CC; kk += 32) {
        v16h af0 = load_A_frag(A + kk, CC, m, g);
        v16h af1 = load_A_frag(A + (size_t)16 * CC + kk, CC, m, g);
        const _Float16* bb = wT + ((size_t)(tn * 16 + m) * 512 + kk + 16 * g);
        v16h bfq = *(const v16h*)(bb);
        v16h bfk = *(const v16h*)(bb + 1 * 262144);
        v16h bfv = *(const v16h*)(bb + 2 * 262144);
        aq0 = wmma_f16(af0, bfq, aq0);
        aq1 = wmma_f16(af1, bfq, aq1);
        ak0 = wmma_f16(af0, bfk, ak0);
        ak1 = wmma_f16(af1, bfk, ak1);
        av0 = wmma_f16(af0, bfv, av0);
        av1 = wmma_f16(af1, bfv, av1);
    }
    float biasq = bq[tn * 16 + m];
    float biask = bk[tn * 16 + m];
    float biasv = bv[tn * 16 + m];
#pragma unroll
    for (int r = 0; r < 8; r++) {
        int tok0 = tp * 32 + r + 8 * g;
        int tok1 = tok0 + 16;
        size_t row0 = ((size_t)b * HWN + tok0) * CC + tn * 16 + m;
        size_t row1 = ((size_t)b * HWN + tok1) * CC + tn * 16 + m;
        qh[row0] = (_Float16)(aq0[r] + biasq);
        qh[row1] = (_Float16)(aq1[r] + biasq);
        kh[row0] = (_Float16)(ak0[r] + biask);
        kh[row1] = (_Float16)(ak1[r] + biask);
        size_t vrow = ((size_t)b * CC + tn * 16 + m) * HWN;
        vT[vrow + tok0] = (_Float16)(av0[r] + biasv);
        vT[vrow + tok1] = (_Float16)(av1[r] + biasv);
    }
}

// ---- 5. Flash pass A: 2 query tiles/wave, per-row softmax max & sum ---------
__global__ void attn_stats(const _Float16* __restrict__ qh, const _Float16* __restrict__ kh,
                           float* __restrict__ Mws, float* __restrict__ Lws) {
    int wid  = blockIdx.x * 8 + (threadIdx.x >> 5);    // 512 waves
    int lane = threadIdx.x & 31;
    int m = lane & 15, g = lane >> 4;
    int qp = wid & 127;                                // 32-row query block
    int b  = wid >> 7;
    const _Float16* Q = qh + ((size_t)b * HWN + qp * 32) * CC;
    const _Float16* K = kh + (size_t)b * HWN * CC;
    v16h qa0[16], qa1[16];
#pragma unroll
    for (int i = 0; i < 16; i++) {
        qa0[i] = load_A_frag(Q + i * 32, CC, m, g);
        qa1[i] = load_A_frag(Q + (size_t)16 * CC + i * 32, CC, m, g);
    }
    float m0[8], l0[8], m1[8], l1[8];
#pragma unroll
    for (int r = 0; r < 8; r++) {
        m0[r] = -3.0e38f; l0[r] = 0.f;
        m1[r] = -3.0e38f; l1[r] = 0.f;
    }
    for (int j0 = 0; j0 < HWN; j0 += 16) {
        v8f s0 = {}, s1 = {};
        const _Float16* Kb = K + (size_t)(j0 + m) * CC + 16 * g;
#pragma unroll
        for (int i = 0; i < 16; i++) {
            v16h bf = *(const v16h*)(Kb + i * 32);
            s0 = wmma_f16(qa0[i], bf, s0);
            s1 = wmma_f16(qa1[i], bf, s1);
        }
        online_update(s0, m0, l0);
        online_update(s1, m1, l1);
    }
    if ((lane & 15) == 0) {
#pragma unroll
        for (int r = 0; r < 8; r++) {
            int row0 = b * HWN + qp * 32 + r + 8 * g;
            Mws[row0] = m0[r];       Lws[row0] = l0[r];
            Mws[row0 + 16] = m1[r];  Lws[row0 + 16] = l1[r];
        }
    }
}

// ---- 6. Flash pass B: 2 query tiles/wave, recompute scores, P via LDS, P*V --
__global__ void attn_out(const _Float16* __restrict__ qh, const _Float16* __restrict__ kh,
                         const _Float16* __restrict__ vT, const float* __restrict__ Mws,
                         const float* __restrict__ Lws, _Float16* __restrict__ oh) {
    __shared__ __align__(64) _Float16 Pb[8][2][512];
    int wv   = threadIdx.x >> 5;
    int wid  = blockIdx.x * 8 + wv;                    // 4096 waves
    int lane = threadIdx.x & 31;
    int m = lane & 15, g = lane >> 4;
    int dc = wid & 7;                                  // 64-channel value chunk
    int qp = (wid >> 3) & 127;                         // 32-row query block
    int b  = wid >> 10;
    const _Float16* Q = qh + ((size_t)b * HWN + qp * 32) * CC;
    const _Float16* K = kh + (size_t)b * HWN * CC;
    const _Float16* V = vT + (size_t)b * CC * HWN + (size_t)(dc * 64) * HWN;
    v16h qa0[16], qa1[16];
#pragma unroll
    for (int i = 0; i < 16; i++) {
        qa0[i] = load_A_frag(Q + i * 32, CC, m, g);
        qa1[i] = load_A_frag(Q + (size_t)16 * CC + i * 32, CC, m, g);
    }
    float m0[8], li0[8], m1[8], li1[8];
#pragma unroll
    for (int r = 0; r < 8; r++) {
        int row0 = b * HWN + qp * 32 + r + 8 * g;
        m0[r] = Mws[row0];       li0[r] = 1.f / Lws[row0];
        m1[r] = Mws[row0 + 16];  li1[r] = 1.f / Lws[row0 + 16];
    }
    v8f o0[4], o1[4];
#pragma unroll
    for (int t = 0; t < 4; t++) { o0[t] = (v8f){}; o1[t] = (v8f){}; }
    _Float16* P0 = &Pb[wv][0][0];
    _Float16* P1 = &Pb[wv][1][0];
    for (int j0 = 0; j0 < HWN; j0 += 32) {
        v8f sa = {}, sb = {}, sc = {}, sd = {};
        const _Float16* Kb0 = K + (size_t)(j0 + m) * CC + 16 * g;
        const _Float16* Kb1 = Kb0 + (size_t)16 * CC;
#pragma unroll
        for (int i = 0; i < 16; i++) {
            v16h b0 = *(const v16h*)(Kb0 + i * 32);
            v16h b1 = *(const v16h*)(Kb1 + i * 32);
            sa = wmma_f16(qa0[i], b0, sa);
            sb = wmma_f16(qa0[i], b1, sb);
            sc = wmma_f16(qa1[i], b0, sc);
            sd = wmma_f16(qa1[i], b1, sd);
        }
#pragma unroll
        for (int r = 0; r < 8; r++) {
            int rr = (r + 8 * g) * 32;
            P0[rr + m]      = (_Float16)__expf(sa[r] * SCALE - m0[r]);
            P0[rr + 16 + m] = (_Float16)__expf(sb[r] * SCALE - m0[r]);
            P1[rr + m]      = (_Float16)__expf(sc[r] * SCALE - m1[r]);
            P1[rr + 16 + m] = (_Float16)__expf(sd[r] * SCALE - m1[r]);
        }
        asm volatile("s_wait_dscnt 0" ::: "memory");   // wave-private LDS tile ready
        v16h pa0 = load_A_frag(P0, 32, m, g);
        v16h pa1 = load_A_frag(P1, 32, m, g);
        const _Float16* Vb = V + (size_t)m * HWN + j0 + 16 * g;
#pragma unroll
        for (int t = 0; t < 4; t++) {
            v16h bf = *(const v16h*)(Vb + (size_t)(16 * t) * HWN);
            o0[t] = wmma_f16(pa0, bf, o0[t]);
            o1[t] = wmma_f16(pa1, bf, o1[t]);
        }
    }
#pragma unroll
    for (int t = 0; t < 4; t++)
#pragma unroll
        for (int r = 0; r < 8; r++) {
            int tok0 = qp * 32 + r + 8 * g;
            int ch   = dc * 64 + 16 * t + m;
            oh[((size_t)b * HWN + tok0) * CC + ch]      = (_Float16)(o0[t][r] * li0[r]);
            oh[((size_t)b * HWN + tok0 + 16) * CC + ch] = (_Float16)(o1[t][r] * li1[r]);
        }
}

// ---- 7. Output projection + bias + residual: 4 token tiles per wave ---------
__global__ void out_gemm(const _Float16* __restrict__ oh, const _Float16* __restrict__ woT,
                         const float* __restrict__ bo, const float* __restrict__ x,
                         float* __restrict__ out) {
    int wid  = blockIdx.x * 8 + (threadIdx.x >> 5);    // 8192 waves
    int lane = threadIdx.x & 31;
    int m = lane & 15, g = lane >> 4;
    int tn = wid & 31;               // out-channel tile
    int tq = (wid >> 5) & 63;        // 64-row token block
    int b  = wid >> 11;
    const _Float16* A = oh + ((size_t)b * HWN + tq * 64) * CC;
    v8f acc[4];
#pragma unroll
    for (int t = 0; t < 4; t++) acc[t] = (v8f){};
    for (int kk = 0; kk < CC; kk += 32) {
        v16h bf = *(const v16h*)(woT + (size_t)(tn * 16 + m) * 512 + kk + 16 * g);
#pragma unroll
        for (int t = 0; t < 4; t++) {
            v16h af = load_A_frag(A + (size_t)(t * 16) * CC + kk, CC, m, g);
            acc[t] = wmma_f16(af, bf, acc[t]);
        }
    }
    float bias = bo[tn * 16 + m];
#pragma unroll
    for (int t = 0; t < 4; t++)
#pragma unroll
        for (int r = 0; r < 8; r++) {
            size_t idx = ((size_t)b * HWN + tq * 64 + t * 16 + r + 8 * g) * CC + tn * 16 + m;
            out[idx] = acc[t][r] + bias + x[idx];
        }
}

// ---- launch -----------------------------------------------------------------
extern "C" void kernel_launch(void* const* d_in, const int* in_sizes, int n_in,
                              void* d_out, int out_size, void* d_ws, size_t ws_size,
                              hipStream_t stream) {
    (void)in_sizes; (void)n_in; (void)out_size; (void)ws_size;
    const float* x     = (const float*)d_in[0];
    const float* gamma = (const float*)d_in[1];
    const float* beta  = (const float*)d_in[2];
    const float* wq    = (const float*)d_in[3];
    const float* bq    = (const float*)d_in[4];
    const float* wk    = (const float*)d_in[5];
    const float* bk    = (const float*)d_in[6];
    const float* wv    = (const float*)d_in[7];
    const float* bv    = (const float*)d_in[8];
    const float* wo    = (const float*)d_in[9];
    const float* bo    = (const float*)d_in[10];
    float* out = (float*)d_out;

    char* ws = (char*)d_ws;
    size_t off = 0;
    auto alloc = [&](size_t bytes) -> char* {
        char* p = ws + off;
        off += (bytes + 255) & ~(size_t)255;
        return p;
    };
    const size_t NTOK = (size_t)BB * HWN;     // 16384 tokens
    float*    meanr = (float*)   alloc((size_t)BB * NG * 2 * sizeof(float));
    _Float16* xn    = (_Float16*)alloc(NTOK * CC * 2);
    _Float16* qh    = (_Float16*)alloc(NTOK * CC * 2);
    _Float16* kh    = (_Float16*)alloc(NTOK * CC * 2);
    _Float16* vT    = (_Float16*)alloc(NTOK * CC * 2);
    _Float16* oh    = (_Float16*)alloc(NTOK * CC * 2);
    _Float16* wT    = (_Float16*)alloc((size_t)4 * CC * CC * 2);
    float*    Mws   = (float*)   alloc(NTOK * sizeof(float));
    float*    Lws   = (float*)   alloc(NTOK * sizeof(float));

    gn_stats  <<<BB * NG, 256, 0, stream>>>(x, meanr);
    gn_norm   <<<(int)((NTOK * CC) / 256), 256, 0, stream>>>(x, gamma, beta, meanr, xn);
    cvt_w     <<<(4 * CC * CC) / 256, 256, 0, stream>>>(wq, wk, wv, wo, wT);
    qkv_gemm  <<<2048, 256, 0, stream>>>(xn, wT, bq, bk, bv, qh, kh, vT);
    attn_stats<<<64, 256, 0, stream>>>(qh, kh, Mws, Lws);
    attn_out  <<<512, 256, 0, stream>>>(qh, kh, vT, Mws, Lws, oh);
    out_gemm  <<<1024, 256, 0, stream>>>(oh, wT + (size_t)3 * CC * CC, bo, x, out);
}
